// HybridGATTransformer_19490561589448
// MI455X (gfx1250) — matrix operand order repaired and checked by reading, not compiled, not run on hardware
//
#include <hip/hip_runtime.h>
#include <hip/hip_bf16.h>

typedef __attribute__((ext_vector_type(16))) _Float16 v16h;
typedef __attribute__((ext_vector_type(8)))  _Float16 v8h;
typedef __attribute__((ext_vector_type(8)))  float    v8f;

// ---------------------------------------------------------------------------
// helpers
// ---------------------------------------------------------------------------
__device__ __forceinline__ float lrelu(float x, float sl) { return x > 0.f ? x : sl * x; }

// order-preserving monotone encoding of float -> uint (for atomicMax)
__device__ __forceinline__ unsigned fenc(float x) {
    unsigned u = __float_as_uint(x);
    return (u & 0x80000000u) ? ~u : (u | 0x80000000u);
}
__device__ __forceinline__ float fdec(unsigned u) {
    u = (u & 0x80000000u) ? (u & 0x7FFFFFFFu) : ~u;
    return __uint_as_float(u);
}

// CDNA5 16-bit A-matrix fragment (16x32) from an LDS f16 tile, row stride lda.
// Per ISA 7.12.2 the per-lane A elements are two contiguous runs of 8 f16:
//   k = kbase + half*8 + [0..7]   and   k = kbase + 16 + half*8 + [0..7]
// -> two 16-byte LDS vector loads per lane.
__device__ __forceinline__ v16h fragA_lds(const _Float16* A, int lda, int kbase, int lane) {
    const int half = (lane >> 4) & 1, m = lane & 15;
    const _Float16* p = A + m * lda + kbase + half * 8;
    v8h lo = *(const v8h*)p;
    v8h hi = *(const v8h*)(p + 16);
    v16h out;
#pragma unroll
    for (int i = 0; i < 8; ++i) {
        out[i] = lo[i];
        out[8 + i] = hi[i];
    }
    return out;
}

// CDNA5 16-bit B-matrix fragment (32x16) built from a row-major fp32 weight
// matrix W[MOUT][KREAL] in global memory (B[k,n] = W[col0+n][k]).
// Lane n needs k = kbase + half*8 + [0..7] and kbase + 16 + half*8 + [0..7] of one row
// -> 4x b128 loads when KREAL is a multiple of 32; otherwise (GAT1, K=8) a
// compile-time zero-guarded scalar path so K-padding never injects NaNs.
template <int KREAL>
__device__ __forceinline__ v16h fragB_gmem(const float* W, int col0, int kbase, int lane) {
    const int half = (lane >> 4) & 1, n = lane & 15;
    const float* p = W + (size_t)(col0 + n) * KREAL + kbase + (half << 3);
    v16h out;
    if (KREAL % 32 == 0) {
        const float4* q = (const float4*)p;
#pragma unroll
        for (int j = 0; j < 4; ++j) {
            float4 f = q[j + (j >> 1) * 2];
            out[4 * j + 0] = (_Float16)f.x;
            out[4 * j + 1] = (_Float16)f.y;
            out[4 * j + 2] = (_Float16)f.z;
            out[4 * j + 3] = (_Float16)f.w;
        }
    } else {
#pragma unroll
        for (int i = 0; i < 16; ++i) {
            const int k = kbase + (half << 3) + i + (i >= 8 ? 8 : 0);
            out[i] = (k < KREAL) ? (_Float16)p[i + (i >= 8 ? 8 : 0)] : (_Float16)0.f;
        }
    }
    return out;
}

// One-wave GEMM (K == KPAD == KREAL): OUT[16][MOUT] = A(16xK f16 LDS) * W^T
// (+bias); results written into per-wave LDS buffers (f32 and/or f16).
// C/D layout per ISA: lane holds rows v+8*half at column (t*16 + lane&15).
template <int K, int MOUT, bool RELU>
__device__ __forceinline__ void wave_gemm(const _Float16* A, const float* W,
                                          const float* bias, float* outF32, _Float16* outF16,
                                          int lane) {
    const int half = (lane >> 4) & 1, n = lane & 15;
#pragma unroll
    for (int t = 0; t < MOUT / 16; ++t) {
        v8f c = {};
#pragma unroll
        for (int s = 0; s < K / 32; ++s) {
            v16h a = fragA_lds(A, K, 32 * s, lane);
            v16h b = fragB_gmem<K>(W, t * 16, 32 * s, lane);
            c = __builtin_amdgcn_wmma_f32_16x16x32_f16(false, a, false, b, (short)0, c, false, false);
        }
        const float bv = bias ? bias[t * 16 + n] : 0.f;
#pragma unroll
        for (int v = 0; v < 8; ++v) {
            const int r = v + half * 8;
            float val = c[v] + bv;
            if (RELU) val = fmaxf(val, 0.f);
            if (outF32) outF32[r * MOUT + t * 16 + n] = val;
            if (outF16) outF16[r * MOUT + t * 16 + n] = (_Float16)val;
        }
    }
}

// Row-wise LayerNorm of (res + delta) over 64 features; lanes 0-15 each own a
// row. Intra-wave LDS ordering keeps this coherent without barriers.
__device__ __forceinline__ void wave_ln(float* res, const float* delta, const float* g,
                                        const float* b, _Float16* outA, int lane) {
    if (lane < 16) {
        const int r = lane;
        float mean = 0.f;
        for (int c = 0; c < 64; ++c) {
            float v = res[r * 64 + c] + delta[r * 64 + c];
            res[r * 64 + c] = v;
            mean += v;
        }
        mean *= (1.f / 64.f);
        float var = 0.f;
        for (int c = 0; c < 64; ++c) {
            float d = res[r * 64 + c] - mean;
            var += d * d;
        }
        var *= (1.f / 64.f);
        const float rstd = rsqrtf(var + 1e-5f);
        for (int c = 0; c < 64; ++c) {
            float o = (res[r * 64 + c] - mean) * rstd * g[c] + b[c];
            res[r * 64 + c] = o;
            if (outA) outA[r * 64 + c] = (_Float16)o;
        }
    }
}

// ---------------------------------------------------------------------------
// kernels
// ---------------------------------------------------------------------------

// h0[n,:8] = x[n,:4] @ fc_feat_w.T + b
__global__ void k_feat(const float* __restrict__ x, const float* __restrict__ w,
                       const float* __restrict__ b, float* __restrict__ h0, int N) {
    int n = blockIdx.x * blockDim.x + threadIdx.x;
    if (n >= N) return;
    float x0 = x[n * 4 + 0], x1 = x[n * 4 + 1], x2 = x[n * 4 + 2], x3 = x[n * 4 + 3];
#pragma unroll
    for (int j = 0; j < 8; ++j)
        h0[(size_t)n * 8 + j] = b[j] + w[j * 4 + 0] * x0 + w[j * 4 + 1] * x1 +
                                w[j * 4 + 2] * x2 + w[j * 4 + 3] * x3;
}

// hfeat = h_in @ W.T  (WMMA; one wave per 16-node tile, 2 waves/block)
template <int KREAL, int KPAD, int MOUT>
__global__ void k_gat_lin(const float* __restrict__ hin, const float* __restrict__ W,
                          float* __restrict__ out, int N) {
    __shared__ __align__(16) _Float16 sA[2][16 * KPAD];
    const int wave = threadIdx.x >> 5, lane = threadIdx.x & 31;
    const int n0 = (blockIdx.x * 2 + wave) * 16;
    if (n0 >= N) return;  // uniform per wave: EXEC stays all-ones for WMMA
    _Float16* A = sA[wave];
    for (int idx = lane; idx < 16 * KPAD; idx += 32) {
        int r = idx / KPAD, k = idx % KPAD;
        int node = n0 + r;
        float v = (node < N && k < KREAL) ? hin[(size_t)node * KREAL + k] : 0.f;
        A[idx] = (_Float16)v;
    }
    const int half = (lane >> 4) & 1, n = lane & 15;
#pragma unroll
    for (int t = 0; t < MOUT / 16; ++t) {
        v8f c = {};
#pragma unroll
        for (int s = 0; s < KPAD / 32; ++s) {
            v16h a = fragA_lds(A, KPAD, 32 * s, lane);
            v16h b = fragB_gmem<KREAL>(W, t * 16, 32 * s, lane);
            c = __builtin_amdgcn_wmma_f32_16x16x32_f16(false, a, false, b, (short)0, c, false, false);
        }
#pragma unroll
        for (int v = 0; v < 8; ++v) {
            int node = n0 + v + half * 8;
            if (node < N) out[(size_t)node * MOUT + t * 16 + n] = c[v];
        }
    }
}

// per-(node,head) attention scores a_s, a_d
template <int C>
__global__ void k_scores(const float* __restrict__ hf, const float* __restrict__ as_w,
                         const float* __restrict__ ad_w, float* __restrict__ sc, int N) {
    int i = blockIdx.x * blockDim.x + threadIdx.x;
    if (i >= N * 4) return;
    int n = i >> 2, h = i & 3;
    const float* row = hf + (size_t)n * 4 * C + h * C;
    float s = 0.f, d = 0.f;
#pragma unroll
    for (int c = 0; c < C; ++c) {
        s += row[c] * as_w[h * C + c];
        d += row[c] * ad_w[h * C + c];
    }
    sc[(size_t)n * 8 + h] = s;
    sc[(size_t)n * 8 + 4 + h] = d;
}

__global__ void k_initmd(unsigned* __restrict__ md, float* __restrict__ dn, int N) {
    int i = blockIdx.x * blockDim.x + threadIdx.x;
    if (i >= N * 4) return;
    int n = i >> 2, h = i & 3;
    md[(size_t)n * 8 + h] = 0x007FFFFFu;  // fenc(-inf)
    dn[(size_t)n * 8 + 4 + h] = 0.f;
}

__global__ void k_zero(float* __restrict__ p, long n) {
    long i = (long)blockIdx.x * blockDim.x + threadIdx.x;
    if (i < n) p[i] = 0.f;
}

// segment max of leaky(a_s[src]+a_d[dst]) over dst
__global__ void k_edge_max(const int* __restrict__ ei, int E, int N,
                           const float* __restrict__ sc, unsigned* __restrict__ md) {
    int e = blockIdx.x * blockDim.x + threadIdx.x;
    int EE = E + N;
    if (e >= EE) return;
    int s = e < E ? ei[e] : (e - E);
    int d = e < E ? ei[E + e] : (e - E);
#pragma unroll
    for (int h = 0; h < 4; ++h) {
        float v = lrelu(sc[(size_t)s * 8 + h] + sc[(size_t)d * 8 + 4 + h], 0.2f);
        atomicMax(&md[(size_t)d * 8 + h], fenc(v));
    }
}

// segment sum of exp(e - m[dst])
__global__ void k_edge_sum(const int* __restrict__ ei, int E, int N,
                           const float* __restrict__ sc, const unsigned* __restrict__ md,
                           float* __restrict__ dn) {
    int e = blockIdx.x * blockDim.x + threadIdx.x;
    int EE = E + N;
    if (e >= EE) return;
    int s = e < E ? ei[e] : (e - E);
    int d = e < E ? ei[E + e] : (e - E);
#pragma unroll
    for (int h = 0; h < 4; ++h) {
        float v = lrelu(sc[(size_t)s * 8 + h] + sc[(size_t)d * 8 + 4 + h], 0.2f);
        float m = fdec(md[(size_t)d * 8 + h]);
        atomicAdd(&dn[(size_t)d * 8 + 4 + h], expf(v - m));
    }
}

// agg[dst] += alpha * hfeat[src]   (one thread per (edge, head))
template <int C>
__global__ void k_edge_agg(const int* __restrict__ ei, int E, int N,
                           const float* __restrict__ sc, const unsigned* __restrict__ md,
                           const float* __restrict__ dn, const float* __restrict__ hf,
                           float* __restrict__ agg) {
    int i = blockIdx.x * blockDim.x + threadIdx.x;
    int EE = E + N;
    if (i >= EE * 4) return;
    int e = i >> 2, h = i & 3;
    int s = e < E ? ei[e] : (e - E);
    int d = e < E ? ei[E + e] : (e - E);
    float v = lrelu(sc[(size_t)s * 8 + h] + sc[(size_t)d * 8 + 4 + h], 0.2f);
    float m = fdec(md[(size_t)d * 8 + h]);
    float alpha = expf(v - m) / dn[(size_t)d * 8 + 4 + h];
    const float* hs = hf + (size_t)s * 4 * C + h * C;
    float* ad = agg + (size_t)d * 4 * C + h * C;
#pragma unroll
    for (int c = 0; c < C; ++c) atomicAdd(&ad[c], alpha * hs[c]);
}

// in-place: a = leaky_relu(a + bias, 0.01)
__global__ void k_epi(float* __restrict__ a, const float* __restrict__ bias, long total, int M) {
    long i = (long)blockIdx.x * blockDim.x + threadIdx.x;
    if (i >= total) return;
    int c = (int)(i % M);
    a[i] = lrelu(a[i] + bias[c], 0.01f);
}

// fused transformer layer (seq-len 1): up to 2 attention blocks (v->out->LN)
// then FFN 64->256(relu)->64 -> LN. In-place on y[N,64]. One wave per tile.
__global__ void k_xf_layer(float* __restrict__ y,
                           const float* vw0, const float* vb0, const float* ow0, const float* ob0,
                           const float* l0g, const float* l0b,
                           const float* vw1, const float* vb1, const float* ow1, const float* ob1,
                           const float* l1g, const float* l1b,
                           const float* f1w, const float* f1b, const float* f2w, const float* f2b,
                           const float* lfg, const float* lfb, int nattn, int N) {
    __shared__ __align__(16) _Float16 sMain[2][16 * 256];
    __shared__ __align__(16) _Float16 sY[2][16 * 64];
    __shared__ __align__(16) float    sRes[2][16 * 64];
    __shared__ __align__(16) float    sTmp[2][16 * 64];
    const int wave = threadIdx.x >> 5, lane = threadIdx.x & 31;
    const int n0 = (blockIdx.x * 2 + wave) * 16;
    if (n0 >= N) return;
    _Float16* Am = sMain[wave];
    _Float16* Ay = sY[wave];
    float* res = sRes[wave];
    float* tmp = sTmp[wave];

    for (int idx = lane; idx < 1024; idx += 32) {
        int r = idx >> 6, c = idx & 63;
        int node = n0 + r;
        float v = (node < N) ? y[(size_t)node * 64 + c] : 0.f;
        res[idx] = v;
        Ay[idx] = (_Float16)v;
    }

    for (int a = 0; a < nattn; ++a) {
        const float* vw = a ? vw1 : vw0;
        const float* vb = a ? vb1 : vb0;
        const float* ow = a ? ow1 : ow0;
        const float* ob = a ? ob1 : ob0;
        const float* lg = a ? l1g : l0g;
        const float* lb = a ? l1b : l0b;
        wave_gemm<64, 64, false>(Ay, vw, vb, nullptr, Am, lane);   // v = y @ vw.T + vb
        wave_gemm<64, 64, false>(Am, ow, ob, tmp, nullptr, lane);  // a = v @ ow.T + ob
        wave_ln(res, tmp, lg, lb, Ay, lane);                       // y = LN(y + a)
    }
    wave_gemm<64, 256, true>(Ay, f1w, f1b, nullptr, Am, lane);     // relu(y@f1.T+b1)
    wave_gemm<256, 64, false>(Am, f2w, f2b, tmp, nullptr, lane);   // @f2.T+b2
    wave_ln(res, tmp, lfg, lfb, nullptr, lane);                    // y = LN(y + f)

    for (int idx = lane; idx < 1024; idx += 32) {
        int r = idx >> 6, c = idx & 63;
        int node = n0 + r;
        if (node < N) y[(size_t)node * 64 + c] = res[idx];
    }
}

// segment sum by graph for mean pool
__global__ void k_pool(const float* __restrict__ y, const int* __restrict__ batch,
                       float* __restrict__ pool, float* __restrict__ cnt, int N) {
    int n = blockIdx.x * blockDim.x + threadIdx.x;
    if (n >= N) return;
    int g = batch[n];
#pragma unroll 8
    for (int c = 0; c < 64; ++c) atomicAdd(&pool[(size_t)g * 64 + c], y[(size_t)n * 64 + c]);
    atomicAdd(&cnt[g], 1.f);
}

// out[g, j] = (pool[g]/max(cnt,1)) @ fc_w.T + fc_b
__global__ void k_fc(const float* __restrict__ pool, const float* __restrict__ cnt,
                     const float* __restrict__ w, const float* __restrict__ b,
                     float* __restrict__ out, int G) {
    int i = blockIdx.x * blockDim.x + threadIdx.x;
    if (i >= G * 33) return;
    int g = i / 33, j = i % 33;
    float inv = 1.f / fmaxf(cnt[g], 1.f);
    float acc = b[j];
    for (int c = 0; c < 64; ++c) acc += pool[(size_t)g * 64 + c] * inv * w[j * 64 + c];
    out[i] = acc;
}

// ---------------------------------------------------------------------------
// launch
// ---------------------------------------------------------------------------
static inline int cdiv(long a, long b) { return (int)((a + b - 1) / b); }

extern "C" void kernel_launch(void* const* d_in, const int* in_sizes, int n_in,
                              void* d_out, int out_size, void* d_ws, size_t ws_size,
                              hipStream_t stream) {
    (void)n_in; (void)ws_size;
    const float* x     = (const float*)d_in[0];
    const int*   ei    = (const int*)d_in[1];
    const int*   batch = (const int*)d_in[2];
    const float* fcw   = (const float*)d_in[3];
    const float* fcb   = (const float*)d_in[4];
    const float* g1w = (const float*)d_in[5],  *g1as = (const float*)d_in[6];
    const float* g1ad = (const float*)d_in[7], *g1b = (const float*)d_in[8];
    const float* g2w = (const float*)d_in[9],  *g2as = (const float*)d_in[10];
    const float* g2ad = (const float*)d_in[11], *g2b = (const float*)d_in[12];
    const float* enc_in_w = (const float*)d_in[13], *enc_in_b = (const float*)d_in[14];
    const float* enc_out_w = (const float*)d_in[15], *enc_out_b = (const float*)d_in[16];
    const float* enc_f1_w = (const float*)d_in[17], *enc_f1_b = (const float*)d_in[18];
    const float* enc_f2_w = (const float*)d_in[19], *enc_f2_b = (const float*)d_in[20];
    const float* enc_l1g = (const float*)d_in[21], *enc_l1b = (const float*)d_in[22];
    const float* enc_l2g = (const float*)d_in[23], *enc_l2b = (const float*)d_in[24];
    const float* sa_in_w = (const float*)d_in[25], *sa_in_b = (const float*)d_in[26];
    const float* sa_out_w = (const float*)d_in[27], *sa_out_b = (const float*)d_in[28];
    const float* ca_in_w = (const float*)d_in[29], *ca_in_b = (const float*)d_in[30];
    const float* ca_out_w = (const float*)d_in[31], *ca_out_b = (const float*)d_in[32];
    const float* de_f1_w = (const float*)d_in[33], *de_f1_b = (const float*)d_in[34];
    const float* de_f2_w = (const float*)d_in[35], *de_f2_b = (const float*)d_in[36];
    const float* de_l1g = (const float*)d_in[37], *de_l1b = (const float*)d_in[38];
    const float* de_l2g = (const float*)d_in[39], *de_l2b = (const float*)d_in[40];
    const float* de_l3g = (const float*)d_in[41], *de_l3b = (const float*)d_in[42];
    const float* fc_w = (const float*)d_in[43], *fc_b = (const float*)d_in[44];

    const int N = in_sizes[0] / 4;
    const int E = in_sizes[1] / 2;
    const int G = out_size / 33;
    const int EE = E + N;

    float* ws = (float*)d_ws;
    float*    h0   = ws;                     // 8N
    float*    sc   = ws + 8L * N;            // 8N: a_s[4] | a_d[4] per node
    unsigned* mdp  = (unsigned*)(ws + 16L * N);  // [N][8]: m-enc[4] | denom[4]
    float*    dnp  = ws + 16L * N;           // same buffer as float
    float*    hf   = ws + 24L * N;           // 128N (hfeat1; later hfeat2|agg2)
    float*    agg  = ws + 152L * N;          // 128N (agg1 -> h1; later pool)
    float*    yv   = hf + 64L * N;           // y lives in upper half of hf region
    float*    pool = agg;                    // 64*G, free after GAT2 linear
    float*    cnt  = agg + 64L * G;

    const int T = (N + 15) / 16;
    const int tileBlocks = (T + 1) / 2;  // 2 waves/block, 1 tile/wave

    // ---- feature fc + GAT layer 1 ----
    k_feat<<<cdiv(N, 256), 256, 0, stream>>>(x, fcw, fcb, h0, N);
    k_gat_lin<8, 32, 128><<<tileBlocks, 64, 0, stream>>>(h0, g1w, hf, N);
    k_scores<32><<<cdiv(4L * N, 256), 256, 0, stream>>>(hf, g1as, g1ad, sc, N);
    k_initmd<<<cdiv(4L * N, 256), 256, 0, stream>>>(mdp, dnp, N);
    k_zero<<<cdiv(128L * N, 256), 256, 0, stream>>>(agg, 128L * N);
    k_edge_max<<<cdiv(EE, 256), 256, 0, stream>>>(ei, E, N, sc, mdp);
    k_edge_sum<<<cdiv(EE, 256), 256, 0, stream>>>(ei, E, N, sc, mdp, dnp);
    k_edge_agg<32><<<cdiv(4L * EE, 256), 256, 0, stream>>>(ei, E, N, sc, mdp, dnp, hf, agg);
    k_epi<<<cdiv(128L * N, 256), 256, 0, stream>>>(agg, g1b, 128L * N, 128);  // agg -> h1

    // ---- GAT layer 2 (hfeat2 reuses hf[0..64N), agg2 = y region) ----
    k_gat_lin<128, 128, 64><<<tileBlocks, 64, 0, stream>>>(agg, g2w, hf, N);
    k_scores<16><<<cdiv(4L * N, 256), 256, 0, stream>>>(hf, g2as, g2ad, sc, N);
    k_initmd<<<cdiv(4L * N, 256), 256, 0, stream>>>(mdp, dnp, N);
    k_zero<<<cdiv(64L * N, 256), 256, 0, stream>>>(yv, 64L * N);
    k_edge_max<<<cdiv(EE, 256), 256, 0, stream>>>(ei, E, N, sc, mdp);
    k_edge_sum<<<cdiv(EE, 256), 256, 0, stream>>>(ei, E, N, sc, mdp, dnp);
    k_edge_agg<16><<<cdiv(4L * EE, 256), 256, 0, stream>>>(ei, E, N, sc, mdp, dnp, hf, yv);
    k_epi<<<cdiv(64L * N, 256), 256, 0, stream>>>(yv, g2b, 64L * N, 64);  // agg2 -> y

    // ---- transformer: 2 encoder layers (1 attn) + 2 decoder layers (2 attn) ----
    for (int i = 0; i < 2; ++i) {
        const float* vw = enc_in_w + (size_t)i * 192 * 64 + 128 * 64;
        const float* vb = enc_in_b + (size_t)i * 192 + 128;
        k_xf_layer<<<tileBlocks, 64, 0, stream>>>(
            yv, vw, vb, enc_out_w + (size_t)i * 64 * 64, enc_out_b + (size_t)i * 64,
            enc_l1g + (size_t)i * 64, enc_l1b + (size_t)i * 64,
            vw, vb, enc_out_w + (size_t)i * 64 * 64, enc_out_b + (size_t)i * 64,  // unused
            enc_l1g + (size_t)i * 64, enc_l1b + (size_t)i * 64,                   // unused
            enc_f1_w + (size_t)i * 256 * 64, enc_f1_b + (size_t)i * 256,
            enc_f2_w + (size_t)i * 64 * 256, enc_f2_b + (size_t)i * 64,
            enc_l2g + (size_t)i * 64, enc_l2b + (size_t)i * 64, 1, N);
    }
    for (int i = 0; i < 2; ++i) {
        const float* svw = sa_in_w + (size_t)i * 192 * 64 + 128 * 64;
        const float* svb = sa_in_b + (size_t)i * 192 + 128;
        const float* cvw = ca_in_w + (size_t)i * 192 * 64 + 128 * 64;
        const float* cvb = ca_in_b + (size_t)i * 192 + 128;
        k_xf_layer<<<tileBlocks, 64, 0, stream>>>(
            yv, svw, svb, sa_out_w + (size_t)i * 64 * 64, sa_out_b + (size_t)i * 64,
            de_l1g + (size_t)i * 64, de_l1b + (size_t)i * 64,
            cvw, cvb, ca_out_w + (size_t)i * 64 * 64, ca_out_b + (size_t)i * 64,
            de_l2g + (size_t)i * 64, de_l2b + (size_t)i * 64,
            de_f1_w + (size_t)i * 256 * 64, de_f1_b + (size_t)i * 256,
            de_f2_w + (size_t)i * 64 * 256, de_f2_b + (size_t)i * 64,
            de_l3g + (size_t)i * 64, de_l3b + (size_t)i * 64, 2, N);
    }

    // ---- mean pool + final fc ----
    k_zero<<<cdiv(64L * G + G, 256), 256, 0, stream>>>(pool, 64L * G + G);
    k_pool<<<cdiv(N, 256), 256, 0, stream>>>(yv, batch, pool, cnt, N);
    k_fc<<<cdiv(33L * G, 256), 256, 0, stream>>>(pool, cnt, fc_w, fc_b, (float*)d_out, G);
}
